// Attention_Fusion_54623394071054
// MI455X (gfx1250) — matrix operand order repaired and checked by reading, not compiled, not run on hardware
//
#include <hip/hip_runtime.h>
#include <stdint.h>

// Problem constants (from reference)
#define DIMC   1024
#define NHEADS 16
#define HDIM   64
#define SEQ    2048
#define BATCH  2
#define NTOK   (BATCH*SEQ)            // 4096 tokens
#define QSCALE_L2E 0.1803368801f      // 64^-0.5 * log2(e): softmax in exp2 domain

// ---------------------------------------------------------------------------
// CDNA5 building blocks
// ---------------------------------------------------------------------------
typedef __attribute__((ext_vector_type(16))) __bf16 bf16x16;  // WMMA A/B frag
typedef __attribute__((ext_vector_type(8)))  float  f32x8;    // WMMA C/D frag

struct __attribute__((aligned(16))) U128 { unsigned int u[4]; };
struct __attribute__((aligned(16))) F4   { float f[4]; };
struct __attribute__((aligned(16))) B8   { __bf16 b[8]; };

union BFrag { bf16x16 v; U128 q[2]; };

__device__ inline f32x8 zero8() { f32x8 z;
#pragma unroll
  for (int i = 0; i < 8; ++i) z[i] = 0.0f; return z; }

__device__ inline f32x8 wmma_bf16(bf16x16 a, bf16x16 b, f32x8 c) {
  // v_wmma_f32_16x16x32_bf16  D = A(16x32) x B(32x16) + C
  return __builtin_amdgcn_wmma_f32_16x16x32_bf16(false, a, false, b,
                                                 (short)0, c, false, false);
}

#if __has_builtin(__builtin_amdgcn_s_wait_asynccnt)
#define WAIT_ASYNC(n) __builtin_amdgcn_s_wait_asynccnt(n)
#else
#define WAIT_ASYNC(n) asm volatile("s_wait_asynccnt %0" :: "i"(n) : "memory")
#endif

// Async global->LDS 16B copy (gfx1250). VDST VGPR holds the LDS byte address
// (low 32 bits of the flat shared pointer ARE the LDS address per ISA 10.2).
__device__ inline void async_cp16(void* lds, const void* gptr) {
  unsigned laddr = (unsigned)(uintptr_t)lds;
  unsigned long long gaddr = (unsigned long long)(uintptr_t)gptr;
  asm volatile("global_load_async_to_lds_b128 %0, %1, off"
               :: "v"(laddr), "v"(gaddr) : "memory");
}

// ---------------------------------------------------------------------------
// Kernel 0: one-shot fp32 -> bf16 convert (x and the four weight matrices).
// Removes all per-GEMM-block cvt VALU work and halves GEMM HBM traffic.
// ---------------------------------------------------------------------------
__global__ __launch_bounds__(256) void cvt_kernel(
    const float* __restrict__ src, __bf16* __restrict__ dst, int n)
{
  int i = (blockIdx.x * 256 + threadIdx.x) * 8;
  if (i < n) {
    F4 a = *(const F4*)&src[i];
    F4 b = *(const F4*)&src[i + 4];
    B8 o;
#pragma unroll
    for (int e = 0; e < 4; ++e) { o.b[e] = (__bf16)a.f[e]; o.b[4 + e] = (__bf16)b.f[e]; }
    *(B8*)&dst[i] = o;
  }
}

// ---------------------------------------------------------------------------
// Shared bf16 GEMM mainloop: C(128x128) += A(128xK) * B(128xK)^T, K=DIMC.
// Both operands are bf16 row-major-along-K; tiles streamed by async-to-LDS
// with double buffering; 8 waves, each wave owns a 32x64 slab (8 wmma/step).
// ---------------------------------------------------------------------------
__device__ __forceinline__ void gemm_bf16_128x128(
    const __bf16* __restrict__ Ag,   // row mbase of A, stride DIMC
    const __bf16* __restrict__ Bg,   // row nbase of B, stride DIMC
    __bf16 (&As)[2][128 * 32], __bf16 (&Bs)[2][128 * 32],
    f32x8 (&acc)[2][4])
{
  const int tid = threadIdx.x, lane = tid & 31;
  const int wid = tid >> 5, hi = lane >> 4, lr = lane & 15;
  const int Moff = (wid & 3) * 32, Noff = (wid >> 2) * 64;

  auto issue = [&](int k0, int buf) {
#pragma unroll
    for (int it = 0; it < 2; ++it) {
      int g = tid + it * 256, r = g >> 2, c = (g & 3) * 8;  // 128 rows x 4 chunks
      async_cp16(&As[buf][r * 32 + c], &Ag[(size_t)r * DIMC + k0 + c]);
      async_cp16(&Bs[buf][r * 32 + c], &Bg[(size_t)r * DIMC + k0 + c]);
    }
  };

  issue(0, 0);
  const int KT = DIMC / 32;
  for (int kk = 0; kk < KT; ++kk) {
    const int cur = kk & 1;
    if (kk + 1 < KT) { issue((kk + 1) * 32, cur ^ 1); WAIT_ASYNC(4); }
    else             { WAIT_ASYNC(0); }
    __syncthreads();

    const __bf16* A  = As[cur];
    const __bf16* Bt = Bs[cur];
    BFrag af[2], bf[4];
#pragma unroll
    for (int mt = 0; mt < 2; ++mt) {
      int r = Moff + mt * 16 + lr;
      af[mt].q[0] = *(const U128*)&A[r * 32 + hi * 8];
      af[mt].q[1] = *(const U128*)&A[r * 32 + 16 + hi * 8];
    }
#pragma unroll
    for (int nt = 0; nt < 4; ++nt) {
      int r = Noff + nt * 16 + lr;
      bf[nt].q[0] = *(const U128*)&Bt[r * 32 + hi * 8];
      bf[nt].q[1] = *(const U128*)&Bt[r * 32 + 16 + hi * 8];
    }
#pragma unroll
    for (int mt = 0; mt < 2; ++mt)
#pragma unroll
      for (int nt = 0; nt < 4; ++nt)
        acc[mt][nt] = wmma_bf16(af[mt].v, bf[nt].v, acc[mt][nt]);

    __syncthreads();   // protect buffers before next iteration's async issue
  }
}

// ---------------------------------------------------------------------------
// Kernel 1: fused QKV projection (bf16 x bf16 -> f32 acc).
// grid.y selects q/k/v (3 x 8 n-tiles). Epilogue: Q -> [B,H,N,Dh] pre-scaled
// by 64^-0.5*log2e; K -> [B,H,N,Dh]; V stored transposed [B,H,Dh,N].
// ---------------------------------------------------------------------------
__global__ __launch_bounds__(256) void qkv_kernel(
    const __bf16* __restrict__ xb,
    const __bf16* __restrict__ Wqb, const float* __restrict__ bq,
    const __bf16* __restrict__ Wkb, const float* __restrict__ bk,
    const __bf16* __restrict__ Wvb, const float* __restrict__ bv,
    __bf16* __restrict__ Qo, __bf16* __restrict__ Ko, __bf16* __restrict__ Vto)
{
  __shared__ __bf16 As[2][128 * 32];
  __shared__ __bf16 Bs[2][128 * 32];

  const int tid = threadIdx.x, lane = tid & 31;
  const int wid = tid >> 5, hi = lane >> 4, lr = lane & 15;
  const int mbase = blockIdx.x * 128;
  const int sel   = blockIdx.y >> 3;            // 0=q 1=k 2=v
  const int nbase = (blockIdx.y & 7) * 128;
  const __bf16* W   = sel == 0 ? Wqb : (sel == 1 ? Wkb : Wvb);
  const float* bias = sel == 0 ? bq  : (sel == 1 ? bk  : bv);
  const int Moff = (wid & 3) * 32, Noff = (wid >> 2) * 64;

  f32x8 acc[2][4];
#pragma unroll
  for (int i = 0; i < 2; ++i)
#pragma unroll
    for (int j = 0; j < 4; ++j) acc[i][j] = zero8();

  gemm_bf16_128x128(xb + (size_t)mbase * DIMC, W + (size_t)nbase * DIMC,
                    As, Bs, acc);

  // epilogue: C layout -> lane holds col n, rows m = vgpr (+8 for upper half)
#pragma unroll
  for (int mt = 0; mt < 2; ++mt)
#pragma unroll
    for (int nt = 0; nt < 4; ++nt) {
      int nloc = nbase + Noff + nt * 16 + lr;    // 0..1023 within this output
      int h = nloc >> 6, dh = nloc & 63;
      float bv_ = bias[nloc];
      int m0 = mbase + Moff + mt * 16 + hi * 8;
      int b_ = m0 >> 11, pos0 = m0 & 2047;
      if (sel == 2) {                            // V^T: pos contiguous -> b128 store
        B8 pk;
#pragma unroll
        for (int j = 0; j < 8; ++j) pk.b[j] = (__bf16)(acc[mt][nt][j] + bv_);
        *(B8*)&Vto[(((size_t)b_ * NHEADS + h) * HDIM + dh) * SEQ + pos0] = pk;
      } else {
#pragma unroll
        for (int j = 0; j < 8; ++j) {
          float v = acc[mt][nt][j] + bv_;
          size_t idx = (((size_t)b_ * NHEADS + h) * SEQ + (pos0 + j)) * HDIM + dh;
          if (sel == 0) Qo[idx] = (__bf16)(v * QSCALE_L2E);
          else          Ko[idx] = (__bf16)v;
        }
      }
    }
}

// ---------------------------------------------------------------------------
// Kernel 2: flash attention. Grid (N/128, B*H). 8 waves x 16 queries each.
// Computes S^T = K*Q^T and O^T = V^T*P^T so softmax stats are per-lane and
// the epilogue stores are contiguous. K/V tiles staged by async-to-LDS.
// Scores are in log2 domain (Q pre-scaled), softmax uses exp2.
// ---------------------------------------------------------------------------
__global__ __launch_bounds__(256) void attn_kernel(
    const __bf16* __restrict__ Q, const __bf16* __restrict__ K,
    const __bf16* __restrict__ Vt, __bf16* __restrict__ Ao)
{
  __shared__ __bf16 Ks[2][32 * 64];   // [key][dh]
  __shared__ __bf16 Vs[2][64 * 32];   // [dh][key]   (V already transposed)

  const int tid = threadIdx.x, lane = tid & 31;
  const int wid = tid >> 5, hi = lane >> 4, lr = lane & 15;
  const int bh = blockIdx.y, b_ = bh >> 4, h = bh & 15;
  const int q0 = blockIdx.x * 128 + wid * 16;

  const __bf16* Qp = Q  + (((size_t)bh) * SEQ + q0) * HDIM;
  const __bf16* Kp = K  + ((size_t)bh) * SEQ * HDIM;
  const __bf16* Vp = Vt + ((size_t)bh) * HDIM * SEQ;

  // Q^T B-fragments (lane: query = lr, dh = f*32 + hi*8 and f*32 + 16 + hi*8)
  BFrag qf[2];
#pragma unroll
  for (int f = 0; f < 2; ++f) {
    qf[f].q[0] = *(const U128*)&Qp[lr * HDIM + f * 32 + hi * 8];
    qf[f].q[1] = *(const U128*)&Qp[lr * HDIM + f * 32 + 16 + hi * 8];
  }

  f32x8 o[4];
#pragma unroll
  for (int c = 0; c < 4; ++c) o[c] = zero8();
  float mrow = -1e30f, lrow = 0.0f;

  const int kr = tid >> 3, kc = (tid & 7) * 8;   // K tile: 32x64, 16B/thread
  const int vr = tid >> 2, vc = (tid & 3) * 8;   // V tile: 64x32, 16B/thread
  // issue key-block 0
  async_cp16(&Ks[0][kr * 64 + kc], &Kp[(size_t)kr * HDIM + kc]);
  async_cp16(&Vs[0][vr * 32 + vc], &Vp[(size_t)vr * SEQ + vc]);

  const int NB = SEQ / 32;
  for (int jb = 0; jb < NB; ++jb) {
    const int cur = jb & 1;
    if (jb + 1 < NB) {                           // prefetch next key-block
      int j0 = (jb + 1) * 32;
      async_cp16(&Ks[cur ^ 1][kr * 64 + kc], &Kp[((size_t)j0 + kr) * HDIM + kc]);
      async_cp16(&Vs[cur ^ 1][vr * 32 + vc], &Vp[(size_t)vr * SEQ + j0 + vc]);
      WAIT_ASYNC(2);                             // current block's 2 copies done
    } else {
      WAIT_ASYNC(0);
    }
    __syncthreads();

    const __bf16* Kb = Ks[cur];
    const __bf16* Vb = Vs[cur];

    // S^T tiles: (16 keys x 16 queries) x2, contraction over dh=64 (2 wmma)
    f32x8 st[2];
#pragma unroll
    for (int t = 0; t < 2; ++t) {
      BFrag kf;
      kf.q[0] = *(const U128*)&Kb[(t * 16 + lr) * 64 + hi * 8];
      kf.q[1] = *(const U128*)&Kb[(t * 16 + lr) * 64 + 16 + hi * 8];
      st[t] = wmma_bf16(kf.v, qf[0].v, zero8());
      kf.q[0] = *(const U128*)&Kb[(t * 16 + lr) * 64 + 32 + hi * 8];
      kf.q[1] = *(const U128*)&Kb[(t * 16 + lr) * 64 + 48 + hi * 8];
      st[t] = wmma_bf16(kf.v, qf[1].v, st[t]);
    }

    // online softmax (log2 domain): lane q holds 16 scores split with q+16
    float mx = st[0][0];
#pragma unroll
    for (int v = 1; v < 8; ++v) mx = fmaxf(mx, st[0][v]);
#pragma unroll
    for (int v = 0; v < 8; ++v) mx = fmaxf(mx, st[1][v]);
    mx = fmaxf(mx, __shfl_xor(mx, 16));
    float mnew  = fmaxf(mrow, mx);
    float alpha = exp2f(mrow - mnew);
    float p0[8], p1[8], psum = 0.0f;
#pragma unroll
    for (int v = 0; v < 8; ++v) {
      p0[v] = exp2f(st[0][v] - mnew);
      p1[v] = exp2f(st[1][v] - mnew);
      psum += p0[v] + p1[v];
    }
    psum += __shfl_xor(psum, 16);
    lrow = lrow * alpha + psum;
    mrow = mnew;
#pragma unroll
    for (int c = 0; c < 4; ++c) o[c] *= alpha;

    // P^T B-fragment: lane-half exchange packs keys 0..15 / 16..31
    BFrag pf;
#pragma unroll
    for (int v = 0; v < 8; ++v) {
      float t0 = __shfl_xor(p0[v], 16);
      float t1 = __shfl_xor(p1[v], 16);
      pf.v[v]     = (__bf16)(hi == 0 ? p0[v] : t1);
      pf.v[8 + v] = (__bf16)(hi == 0 ? t0 : p1[v]);
    }

    // O^T += V^T(16dh x 32keys) x P^T : 4 wmma (dh chunks)
#pragma unroll
    for (int c = 0; c < 4; ++c) {
      BFrag vf;
      vf.q[0] = *(const U128*)&Vb[(c * 16 + lr) * 32 + hi * 16];
      vf.q[1] = *(const U128*)&Vb[(c * 16 + lr) * 32 + hi * 16 + 8];
      o[c] = wmma_bf16(vf.v, pf.v, o[c]);
    }
    __syncthreads();   // protect buffers before next iteration's async issue
  }

  // epilogue: lane q writes its 8 contiguous dh values per tile (b128 stores)
  float rl = 1.0f / lrow;
  size_t row = (size_t)b_ * SEQ + q0 + lr;
  __bf16* op = Ao + row * DIMC + h * HDIM;
#pragma unroll
  for (int c = 0; c < 4; ++c) {
    B8 pk;
#pragma unroll
    for (int j = 0; j < 8; ++j) pk.b[j] = (__bf16)(o[c][j] * rl);
    *(B8*)&op[c * 16 + hi * 8] = pk;
  }
}

// ---------------------------------------------------------------------------
// Kernel 3: output projection. out(f32) = attn(bf16) @ Wp^T(bf16) + bp.
// ---------------------------------------------------------------------------
__global__ __launch_bounds__(256) void proj_kernel(
    const __bf16* __restrict__ A, const __bf16* __restrict__ Wpb,
    const float* __restrict__ bp, float* __restrict__ out)
{
  __shared__ __bf16 As[2][128 * 32];
  __shared__ __bf16 Bs[2][128 * 32];

  const int tid = threadIdx.x, lane = tid & 31;
  const int wid = tid >> 5, hi = lane >> 4, lr = lane & 15;
  const int mbase = blockIdx.x * 128;
  const int nbase = blockIdx.y * 128;
  const int Moff = (wid & 3) * 32, Noff = (wid >> 2) * 64;

  f32x8 acc[2][4];
#pragma unroll
  for (int i = 0; i < 2; ++i)
#pragma unroll
    for (int j = 0; j < 4; ++j) acc[i][j] = zero8();

  gemm_bf16_128x128(A + (size_t)mbase * DIMC, Wpb + (size_t)nbase * DIMC,
                    As, Bs, acc);

#pragma unroll
  for (int mt = 0; mt < 2; ++mt)
#pragma unroll
    for (int nt = 0; nt < 4; ++nt) {
      int n = nbase + Noff + nt * 16 + lr;
      float bv_ = bp[n];
#pragma unroll
      for (int j = 0; j < 8; ++j) {
        int m = mbase + Moff + mt * 16 + hi * 8 + j;
        out[(size_t)m * DIMC + n] = acc[mt][nt][j] + bv_;
      }
    }
}

// ---------------------------------------------------------------------------
// Launch. ws layout (bf16 elems):
//   xb(4M) | Wqb(1M) | Wkb(1M) | Wvb(1M) | Wpb(1M) | Q(4M) | K(4M) | Vt(4M)
//   | attn(4M)   => 24M elems = 48 MB
// ---------------------------------------------------------------------------
extern "C" void kernel_launch(void* const* d_in, const int* in_sizes, int n_in,
                              void* d_out, int out_size, void* d_ws, size_t ws_size,
                              hipStream_t stream) {
  const float* x  = (const float*)d_in[0];
  const float* Wq = (const float*)d_in[1];
  const float* bq = (const float*)d_in[2];
  const float* Wk = (const float*)d_in[3];
  const float* bk = (const float*)d_in[4];
  const float* Wv = (const float*)d_in[5];
  const float* bv = (const float*)d_in[6];
  const float* Wp = (const float*)d_in[7];
  const float* bp = (const float*)d_in[8];
  float* out = (float*)d_out;

  const size_t tokElems = (size_t)NTOK * DIMC;  // 4,194,304
  const size_t wElems   = (size_t)DIMC * DIMC;  // 1,048,576
  __bf16* xb  = (__bf16*)d_ws;
  __bf16* Wqb = xb  + tokElems;
  __bf16* Wkb = Wqb + wElems;
  __bf16* Wvb = Wkb + wElems;
  __bf16* Wpb = Wvb + wElems;
  __bf16* Qb  = Wpb + wElems;
  __bf16* Kb  = Qb  + tokElems;
  __bf16* Vtb = Kb  + tokElems;
  __bf16* An  = Vtb + tokElems;

  const int tokBlocks = (int)(tokElems / 8 / 256);  // 2048
  const int wBlocks   = (int)(wElems / 8 / 256);    // 512
  cvt_kernel<<<tokBlocks, 256, 0, stream>>>(x,  xb,  (int)tokElems);
  cvt_kernel<<<wBlocks,   256, 0, stream>>>(Wq, Wqb, (int)wElems);
  cvt_kernel<<<wBlocks,   256, 0, stream>>>(Wk, Wkb, (int)wElems);
  cvt_kernel<<<wBlocks,   256, 0, stream>>>(Wv, Wvb, (int)wElems);
  cvt_kernel<<<wBlocks,   256, 0, stream>>>(Wp, Wpb, (int)wElems);

  qkv_kernel<<<dim3(NTOK / 128, 24), 256, 0, stream>>>(
      xb, Wqb, bq, Wkb, bk, Wvb, bv, Qb, Kb, Vtb);
  attn_kernel<<<dim3(SEQ / 128, BATCH * NHEADS), 256, 0, stream>>>(
      Qb, Kb, Vtb, An);
  proj_kernel<<<dim3(NTOK / 128, DIMC / 128), 256, 0, stream>>>(
      An, Wp ? Wpb : Wpb, bp, out);
}
